// BilevelGraphAttnEncoder_31963146617206
// MI455X (gfx1250) — compile-verified, matrix operand order checked by reading, not run on hardware
//
#include <hip/hip_runtime.h>
#include <hip/hip_bf16.h>
#include <math.h>

// ---------------------------------------------------------------------------
// BilevelGraphAttnEncoder for gfx1250 (MI455X), round 3.
// Dense GEMMs -> V_WMMA_F32_16X16X32_BF16 (bf16 in / f32 acc):
//  * NT-templated tile count (no per-tile branches, straight-line inner loop)
//  * double-buffered k-loop: next block's A/B loads issued before current
//    block's wmmas -> loads overlap matrix ops
//  * pointer-increment addressing (uniform strides, no 64-bit shifts/mults)
//  * weights pre-packed to bf16 in exact WMMA B-fragment order (32B/lane)
// Irregular work (graph softmax over unsorted segments, gathers, LN) stays
// fp32 VALU + float atomics.
// ---------------------------------------------------------------------------

#define NRES   5000
#define NRP    5008              // node rows padded to multiple of 16
#define KNB    30
#define NEDGE  (NRES * KNB)      // 150000 (multiple of 16)
#define NATOM  (NRES * 14)       // 70000  (multiple of 16)
#define CS     256
#define CZ     128
#define CA     16
#define NH     4
#define DH     64
#define NP     8
#define NL     4
#define NAA    21
#define NRBF   16
#define EPSF   1e-8f

typedef __attribute__((ext_vector_type(16))) __bf16 v16bf;
typedef __attribute__((ext_vector_type(8)))  float  v8f;

// ------------------------------ small helpers ------------------------------

__device__ inline void vsub(const float* a, const float* b, float* o) {
  o[0]=a[0]-b[0]; o[1]=a[1]-b[1]; o[2]=a[2]-b[2];
}
__device__ inline void vcross(const float* a, const float* b, float* o) {
  o[0]=a[1]*b[2]-a[2]*b[1]; o[1]=a[2]*b[0]-a[0]*b[2]; o[2]=a[0]*b[1]-a[1]*b[0];
}
__device__ inline float vdot(const float* a, const float* b) {
  return a[0]*b[0]+a[1]*b[1]+a[2]*b[2];
}
__device__ inline void vnormeps(float* v) {
  float n = sqrtf(vdot(v, v)) + EPSF;
  v[0]/=n; v[1]/=n; v[2]/=n;
}
__device__ inline void atomicMaxF(float* addr, float val) {
  int* ia = (int*)addr;
  int cur = __float_as_int(*addr);
  while (__int_as_float(cur) < val) {
    int old = atomicCAS(ia, cur, __float_as_int(val));
    if (old == cur) break;
    cur = old;
  }
}

// --------------------------- weight pre-packing ----------------------------
// Pack W[Ktrue x Nw] (fp32) into bf16 fragments: for k-block b, column col,
// lane-half h: 16 bf16 = exactly one lane's v16bf B fragment (32B contiguous).
// Zero-filled past Ktrue.

__global__ void pack_w_kernel(const float* __restrict__ W, __bf16* __restrict__ P,
                              int Ktrue, int Nw, int KB) {
  int t = blockIdx.x * blockDim.x + threadIdx.x;
  int total = KB * Nw * 2;
  if (t >= total) return;
  int half = t & 1;
  int rest = t >> 1;
  int col  = rest % Nw;
  int b    = rest / Nw;
  __bf16* o = P + (size_t)t * 16;
#pragma unroll
  for (int i = 0; i < 8; ++i) {
    int kbase = b * 32 + ((i < 4) ? 2 * i : 16 + 2 * (i - 4)) + half * 8;
#pragma unroll
    for (int bit = 0; bit < 2; ++bit) {
      int k = kbase + bit;
      float f = (k < Ktrue) ? W[(size_t)k * Nw + col] : 0.f;
      o[2 * i + bit] = (__bf16)f;
    }
  }
}

// ------------------------------- GEMM (WMMA) -------------------------------

__device__ inline v16bf load_a_frag(const float* __restrict__ Ar, const int* kb) {
  v16bf av;
#pragma unroll
  for (int i = 0; i < 8; ++i) {
    float2 f = *(const float2*)(Ar + kb[i]);
    av[2 * i]     = (__bf16)f.x;
    av[2 * i + 1] = (__bf16)f.y;
  }
  return av;
}

// C[M x Nw] = act(A @ Wp); one wave computes 16 rows x NT*16 cols.
// A fp32 (lda mult of 32, K-pads zero); Wp packed bf16 fragments.
template <int NT, int RELU>
__global__ void __launch_bounds__(32)
gemm_wmma_kernel(const float* __restrict__ A, int lda,
                 const __bf16* __restrict__ Wp,
                 float* __restrict__ C,
                 int M, int KB, int Nw, int ldc, int col0,
                 int padN, int n0base) {
  const int lane = threadIdx.x & 31;
  const int half = lane >> 4;
  const int r    = lane & 15;
  const int m0   = blockIdx.y * 16;
  const int n0   = n0base + blockIdx.x * 64;
  const int row  = m0 + r;

  int kb[8];
#pragma unroll
  for (int i = 0; i < 8; ++i)
    kb[i] = ((i < 4) ? 2 * i : 16 + 2 * (i - 4)) + half * 8;

  const float* pa = A + (size_t)row * lda;
  const v16bf* wp[NT];
#pragma unroll
  for (int t = 0; t < NT; ++t)
    wp[t] = (const v16bf*)Wp + ((size_t)(n0 + t * 16 + r) * 2 + half);
  const size_t wstep = (size_t)Nw * 2;

  v8f acc[NT];
#pragma unroll
  for (int t = 0; t < NT; ++t) acc[t] = (v8f){};

  // prologue: block 0 loads
  v16bf a_cur = load_a_frag(pa, kb);
  v16bf b_cur[NT];
#pragma unroll
  for (int t = 0; t < NT; ++t) { b_cur[t] = *wp[t]; wp[t] += wstep; }
  pa += 32;

  for (int b = 0; b < KB - 1; ++b) {
    // issue next block's loads before this block's wmmas
    v16bf a_nxt = load_a_frag(pa, kb);
    v16bf b_nxt[NT];
#pragma unroll
    for (int t = 0; t < NT; ++t) { b_nxt[t] = *wp[t]; wp[t] += wstep; }
    pa += 32;
#pragma unroll
    for (int t = 0; t < NT; ++t)
      acc[t] = __builtin_amdgcn_wmma_f32_16x16x32_bf16(
          false, a_cur, false, b_cur[t], (short)0, acc[t], false, false);
    a_cur = a_nxt;
#pragma unroll
    for (int t = 0; t < NT; ++t) b_cur[t] = b_nxt[t];
  }
#pragma unroll
  for (int t = 0; t < NT; ++t)
    acc[t] = __builtin_amdgcn_wmma_f32_16x16x32_bf16(
        false, a_cur, false, b_cur[t], (short)0, acc[t], false, false);

#pragma unroll
  for (int t = 0; t < NT; ++t) {
    int col = n0 + t * 16 + r;
#pragma unroll
    for (int j = 0; j < 8; ++j) {
      int orow = m0 + j + half * 8;
      if (orow >= M) continue;
      float v = acc[t][j];
      if (RELU) v = v > 0.f ? v : 0.f;
      C[(size_t)orow * ldc + col0 + col] = v;
    }
  }
  // zero K-pad output columns [Nw, padN) (done by the last column block)
  if (padN > Nw && n0 + NT * 16 >= Nw && blockIdx.x + 1 == gridDim.x) {
    for (int c = Nw + lane; c < padN; c += 32)
      for (int j = 0; j < 16; ++j) {
        int orow = m0 + j;
        if (orow < M) C[(size_t)orow * ldc + col0 + c] = 0.f;
      }
  }
}

// -------------------------------- LayerNorm --------------------------------

__global__ void ln_kernel(float* __restrict__ x, int ldx,
                          const float* __restrict__ delta, int ldd,
                          const float* __restrict__ mask, int M, int D) {
  int warp = threadIdx.x >> 5, lane = threadIdx.x & 31;
  int row = blockIdx.x * 8 + warp;
  if (row >= M) return;
  float* xr = x + (size_t)row * ldx;
  const float* dr = delta ? delta + (size_t)row * ldd : nullptr;
  float s = 0.f, ss = 0.f;
  for (int i = lane; i < D; i += 32) {
    float v = xr[i] + (dr ? dr[i] : 0.f);
    s += v; ss += v * v;
  }
  for (int o = 16; o; o >>= 1) { s += __shfl_xor(s, o, 32); ss += __shfl_xor(ss, o, 32); }
  float mu  = s / D;
  float var = ss / D - mu * mu;
  float inv = rsqrtf(var + 1e-5f);
  float msc = mask ? mask[row] : 1.f;
  for (int i = lane; i < D; i += 32) {
    float v = xr[i] + (dr ? dr[i] : 0.f);
    xr[i] = (v - mu) * inv * msc;
  }
}

__global__ void fill_kernel(float* p, float v, int n) {
  int i = blockIdx.x * blockDim.x + threadIdx.x;
  if (i < n) p[i] = v;
}

// ------------------------------ prep kernels -------------------------------

__global__ void mask_kernel(const int* res, const int* mgm, const int* noi,
                            float* res_f, float* mgm_f, float* noi_f, int n) {
  int i = blockIdx.x * blockDim.x + threadIdx.x;
  if (i >= n) return;
  res_f[i] = res[i] ? 1.f : 0.f;
  mgm_f[i] = mgm[i] ? 1.f : 0.f;
  noi_f[i] = noi[i] ? 1.f : 0.f;
}

__global__ void bb_kernel(const float* atom14, const float* rot, const float* trans,
                          float* bb4, float* bb_local, int n_nodes) {
  int n = blockIdx.x * blockDim.x + threadIdx.x;
  if (n >= n_nodes) return;
  float a0[3], a1[3], a2[3];
  for (int c = 0; c < 3; ++c) {
    a0[c] = atom14[((size_t)n * 14 + 0) * 3 + c];
    a1[c] = atom14[((size_t)n * 14 + 1) * 3 + c];
    a2[c] = atom14[((size_t)n * 14 + 2) * 3 + c];
  }
  float b[3], cc[3], ax[3], cb[3];
  vsub(a1, a0, b); vsub(a2, a1, cc); vcross(b, cc, ax);
  for (int c = 0; c < 3; ++c)
    cb[c] = -0.58273431f * ax[c] + 0.56802827f * b[c] - 0.54067466f * cc[c] + a1[c];
  float* o = bb4 + (size_t)n * 12;
  for (int c = 0; c < 3; ++c) { o[c] = a0[c]; o[3+c] = a1[c]; o[6+c] = a2[c]; o[9+c] = cb[c]; }
  const float* R = rot + (size_t)n * 9;
  const float* T = trans + (size_t)n * 3;
  for (int p = 0; p < 4; ++p) {
    float v[3] = {o[p*3]-T[0], o[p*3+1]-T[1], o[p*3+2]-T[2]};
    for (int i = 0; i < 3; ++i)       // einsum('nji,npj->npi'): R^T
      bb_local[(size_t)n*12 + p*3 + i] = R[0*3+i]*v[0] + R[1*3+i]*v[1] + R[2*3+i]*v[2];
  }
}

__device__ inline void loadX(const float* atom14, int p, float* v) {
  int node = p / 3, atom = p % 3;
  const float* q = atom14 + ((size_t)node * 14 + atom) * 3;
  v[0]=q[0]; v[1]=q[1]; v[2]=q[2];
}

// node_raw row (stride 64): [cosD(3), sinD(3), mgm, onehot*mgm(21), vec(12), 0-pad]
__global__ void node_raw_kernel(const float* atom14, const float* rot, const float* trans,
                                const int* seq, const float* res_f, const float* mgm_f,
                                float* node_raw, int n_nodes) {
  int n = blockIdx.x * blockDim.x + threadIdx.x;
  if (n >= n_nodes) return;
  float feat[40];
  for (int t = 0; t < 3; ++t) {
    int i = 3 * n + t - 1;
    float D = 0.f;
    if (i >= 0 && i <= 3 * n_nodes - 4) {
      float X0[3],X1[3],X2[3],X3[3],u2[3],u1[3],u0[3],n2[3],n1[3];
      loadX(atom14, i, X0); loadX(atom14, i+1, X1);
      loadX(atom14, i+2, X2); loadX(atom14, i+3, X3);
      vsub(X1,X0,u2); vnormeps(u2);
      vsub(X2,X1,u1); vnormeps(u1);
      vsub(X3,X2,u0); vnormeps(u0);
      vcross(u2,u1,n2); vnormeps(n2);
      vcross(u1,u0,n1); vnormeps(n1);
      float cD = vdot(n2,n1);
      cD = fminf(fmaxf(cD, -1.f + 1e-7f), 1.f - 1e-7f);
      float sg = vdot(u2,n1);
      float sgn = (sg > 0.f) ? 1.f : ((sg < 0.f) ? -1.f : 0.f);
      D = sgn * acosf(cD);
    }
    feat[t]     = cosf(D);
    feat[3 + t] = sinf(D);
  }
  float mg = mgm_f[n];
  feat[6] = mg;
  int sq = seq[n];
  for (int a = 0; a < NAA; ++a) feat[7 + a] = (a == sq) ? mg : 0.f;
  const float* R = rot + (size_t)n * 9;
  const float* T = trans + (size_t)n * 3;
  for (int p = 0; p < 4; ++p) {
    float v[3];
    for (int c = 0; c < 3; ++c) v[c] = atom14[((size_t)n*14 + p)*3 + c] - T[c];
    for (int i = 0; i < 3; ++i)
      feat[28 + p*3 + i] = R[0*3+i]*v[0] + R[1*3+i]*v[1] + R[2*3+i]*v[2];
  }
  float rf = res_f[n];
  float* o = node_raw + (size_t)n * 64;
  for (int c = 0; c < 40; ++c) o[c] = feat[c] * rf;
  for (int c = 40; c < 64; ++c) o[c] = 0.f;
}

__global__ void atom_prep_kernel(const int* seq, const int* a14ex,
                                 const float* mgm_f, const float* atom_tab,
                                 float* araw, float* am_f, int n_nodes) {
  int t = blockIdx.x * blockDim.x + threadIdx.x;
  if (t >= n_nodes * 14) return;
  int n = t / 14, a = t % 14;
  float mg = mgm_f[n];
  float sc = (a >= 4) ? mg : 1.f;
  am_f[t]  = (a >= 4) ? mg : (a14ex[t] ? 1.f : 0.f);
  int sq = seq[n];
  float* o = araw + (size_t)t * 32;
  for (int c = 0; c < 9; ++c)
    o[c] = atom_tab[((size_t)sq * 14 + a) * 9 + c] * sc;
  for (int c = 9; c < 32; ++c) o[c] = 0.f;
}

__global__ void edge_raw_kernel(const int* ei, const int* seq, const int* res,
                                const float* noi_f, const float* bb4,
                                const float* bb_local, const float* rot,
                                const float* trans, float* eraw, float* em_f,
                                int n_edges) {
  int e = blockIdx.x * blockDim.x + threadIdx.x;
  if (e >= n_edges) return;
  int dst = ei[e], src = ei[n_edges + e];
  float em = (res[src] && res[dst]) ? 1.f : 0.f;
  em_f[e] = em;
  float* o = eraw + (size_t)e * 352;
  float ns = noi_f[src], nd = noi_f[dst];
  o[0] = nd; o[1] = ns;
  int ss = seq[src], sd = seq[dst];
  for (int a = 0; a < NAA; ++a) {
    o[2 + a]       = (a == ss) ? (1.f - ns) : 0.f;
    o[2 + NAA + a] = (a == sd) ? (1.f - nd) : 0.f;
  }
  const float* bs = bb4 + (size_t)src * 12;
  const float* bd = bb4 + (size_t)dst * 12;
  const float sigma = (22.f - 2.f) / NRBF;
  for (int j = 0; j < 4; ++j)
    for (int k = 0; k < 4; ++k) {
      float dx = bs[j*3+0]-bd[k*3+0]+EPSF, dy = bs[j*3+1]-bd[k*3+1]+EPSF,
            dz = bs[j*3+2]-bd[k*3+2]+EPSF;
      float d = sqrtf(dx*dx + dy*dy + dz*dz);
      for (int ri = 0; ri < NRBF; ++ri) {
        float mu = 2.f + ri * (20.f / (NRBF - 1));
        float u = (d - mu) / sigma;
        o[44 + (j*4+k)*NRBF + ri] = expf(-u*u);
      }
    }
  float dpos = (float)(dst - src);
  for (int i = 0; i < 8; ++i) {
    float f = expf((float)(2*i) * (-0.5756462732485114f));
    float ang = dpos * f;
    o[300 + i] = cosf(ang);
    o[308 + i] = sinf(ang);
  }
  for (int c = 0; c < 12; ++c) o[316 + c] = bb_local[(size_t)src*12 + c] * (1.f - ns);
  const float* R = rot + (size_t)src * 9;
  const float* T = trans + (size_t)src * 3;
  for (int p = 0; p < 4; ++p) {
    float v[3] = {bd[p*3]-T[0], bd[p*3+1]-T[1], bd[p*3+2]-T[2]};
    for (int i = 0; i < 3; ++i)
      o[328 + p*3 + i] = (R[0*3+i]*v[0] + R[1*3+i]*v[1] + R[2*3+i]*v[2]) * (1.f - nd);
  }
  for (int c = 0; c < 340; ++c) o[c] *= em;
  for (int c = 340; c < 352; ++c) o[c] = 0.f;
}

// --------------------------- attention kernels -----------------------------

__global__ void frame_points_kernel(const float* pts, const float* rot,
                                    const float* trans, float* out, int n_nodes) {
  int t = blockIdx.x * blockDim.x + threadIdx.x;
  if (t >= n_nodes * NH * NP) return;
  int n = t / (NH * NP), idx = t % (NH * NP);
  const float* p = pts + (size_t)n * 96 + idx * 3;
  const float* R = rot + (size_t)n * 9;
  const float* T = trans + (size_t)n * 3;
  for (int i = 0; i < 3; ++i)       // einsum('nij,nhpj->nhpi')
    out[(size_t)n*96 + idx*3 + i] =
        R[i*3+0]*p[0] + R[i*3+1]*p[1] + R[i*3+2]*p[2] + T[i];
}

__global__ void logits_kernel(const float* qb, const float* kb, const float* z,
                              const float* Wb_l, const float* gamma_l,
                              const float* qg, const float* kg, const float* em_f,
                              const int* ei, float* logits, float* mb, int n_edges) {
  int t = blockIdx.x * blockDim.x + threadIdx.x;
  if (t >= n_edges * NH) return;
  int e = t >> 2, h = t & 3;
  int dst = ei[e], src = ei[n_edges + e];
  float qk = 0.f;
  for (int d = 0; d < DH; ++d)
    qk += qb[(size_t)dst*CS + h*DH + d] * kb[(size_t)src*CS + h*DH + d];
  float bias = 0.f;
  for (int c = 0; c < CZ; ++c)
    bias += z[(size_t)e*CZ + c] * Wb_l[c*NH + h];
  float pt = 0.f;
  for (int j = 0; j < NP*3; ++j) {
    float dlt = qg[(size_t)dst*96 + h*24 + j] - kg[(size_t)src*96 + h*24 + j];
    pt += dlt * dlt;
  }
  const float w_l = 0.5773502691896258f;       // sqrt(1/3)
  const float w_pt = 0.16666666666666666f;     // sqrt(2/(9*8))
  float sp = log1pf(expf(gamma_l[h]));
  float lg = qk * (w_l / 8.f) + w_l * bias - 0.5f * w_pt * sp * pt;
  if (em_f[e] == 0.f) lg = -1e9f;
  logits[t] = lg;
  atomicMaxF(&mb[dst*NH + h], lg);
}

__global__ void expsum_kernel(const float* mb, const float* em_f, const int* ei,
                              float* logits, float* db, int n_edges) {
  int t = blockIdx.x * blockDim.x + threadIdx.x;
  if (t >= n_edges * NH) return;
  int e = t >> 2, h = t & 3;
  int dst = ei[e];
  float a = expf(logits[t] - mb[dst*NH + h]) * em_f[e];
  logits[t] = a;
  atomicAdd(&db[dst*NH + h], a);
}

__global__ void attn_accum_kernel(const float* logits, const float* db,
                                  const float* vb, const float* kg,
                                  const int* ei, float* nodecat, float* og,
                                  int n_edges) {
  int t = blockIdx.x * blockDim.x + threadIdx.x;
  if (t >= n_edges * NH) return;
  int e = t >> 2, h = t & 3;
  int dst = ei[e], src = ei[n_edges + e];
  float a = logits[t] / (db[dst*NH + h] + 1e-9f);
  if (a == 0.f) return;
  for (int d = 0; d < DH; ++d)
    atomicAdd(&nodecat[(size_t)dst*384 + h*DH + d], a * vb[(size_t)src*CS + h*DH + d]);
  for (int j = 0; j < NP*3; ++j)
    atomicAdd(&og[(size_t)dst*96 + h*24 + j], a * kg[(size_t)src*96 + h*24 + j]);
}

__global__ void finish_o_kernel(const float* og, const float* rot, const float* trans,
                                float* nodecat, int n_nodes) {
  int t = blockIdx.x * blockDim.x + threadIdx.x;
  if (t >= n_nodes * NH * NP) return;
  int n = t / (NH * NP), idx = t % (NH * NP);
  const float* R = rot + (size_t)n * 9;
  const float* T = trans + (size_t)n * 3;
  float v[3], ol[3];
  for (int c = 0; c < 3; ++c) v[c] = og[(size_t)n*96 + idx*3 + c] - T[c];
  for (int i = 0; i < 3; ++i)
    ol[i] = R[0*3+i]*v[0] + R[1*3+i]*v[1] + R[2*3+i]*v[2];
  for (int i = 0; i < 3; ++i) nodecat[(size_t)n*384 + 256 + idx*3 + i] = ol[i];
  nodecat[(size_t)n*384 + 352 + idx] = sqrtf(ol[0]*ol[0]+ol[1]*ol[1]+ol[2]*ol[2] + EPSF);
}

// ------------------------------ concat kernels -----------------------------

// acat row (stride 288): [af(16), s(256), 0-pad(16)]
__global__ void acat_kernel(const float* af, const float* s, float* acat, int n_nodes) {
  int t = blockIdx.x * blockDim.x + threadIdx.x;
  int total = n_nodes * 14 * 288;
  if (t >= total) return;
  int row = t / 288, c = t % 288;
  int n = row / 14;
  float v;
  if (c < CA)       v = af[(size_t)row * 32 + c];
  else if (c < 272) v = s[(size_t)n * CS + (c - CA)];
  else              v = 0.f;
  acat[t] = v;
}

__global__ void ecat_kernel(const float* z, const float* h, const int* ei,
                            float* ecat, int n_edges) {
  int t = blockIdx.x * blockDim.x + threadIdx.x;
  int total = n_edges * 384;
  if (t >= total) return;
  int e = t / 384, c = t % 384;
  int dst = ei[e], src = ei[n_edges + e];
  float v;
  if (c < 128)      v = z[(size_t)e*CZ + c];
  else if (c < 256) v = h[(size_t)dst*CZ + (c - 128)];
  else              v = h[(size_t)src*CZ + (c - 256)];
  ecat[t] = v;
}

// ------------------------------- host driver -------------------------------

template <int NT>
static inline void gemm_launch_nt(hipStream_t st, const float* A, int lda,
                                  const __bf16* Wp, float* C, int Mm, int KB,
                                  int Nn, int ldc, int col0, int padN,
                                  int n0base, int nblk, int relu) {
  dim3 g(nblk, (Mm + 15) / 16);
  if (relu)
    gemm_wmma_kernel<NT, 1><<<g, 32, 0, st>>>(A, lda, Wp, C, Mm, KB, Nn, ldc,
                                              col0, padN, n0base);
  else
    gemm_wmma_kernel<NT, 0><<<g, 32, 0, st>>>(A, lda, Wp, C, Mm, KB, Nn, ldc,
                                              col0, padN, n0base);
}

static inline void gemm(hipStream_t st, const float* A, int lda, const __bf16* Wp,
                        float* C, int Mm, int Ktrue, int Nn, int ldc, int col0,
                        int padN, int relu) {
  int KB = (Ktrue + 31) / 32;
  int nfull = Nn / 64;
  int rem = (Nn - nfull * 64) / 16;
  if (nfull)
    gemm_launch_nt<4>(st, A, lda, Wp, C, Mm, KB, Nn, ldc, col0, padN, 0, nfull, relu);
  if (rem == 1)
    gemm_launch_nt<1>(st, A, lda, Wp, C, Mm, KB, Nn, ldc, col0, padN, nfull * 64, 1, relu);
  else if (rem == 2)
    gemm_launch_nt<2>(st, A, lda, Wp, C, Mm, KB, Nn, ldc, col0, padN, nfull * 64, 1, relu);
  else if (rem == 3)
    gemm_launch_nt<3>(st, A, lda, Wp, C, Mm, KB, Nn, ldc, col0, padN, nfull * 64, 1, relu);
}
static inline void ln(hipStream_t st, float* x, int ldx, const float* delta, int ldd,
                      const float* mask, int Mm, int Dd) {
  ln_kernel<<<(Mm + 7) / 8, 256, 0, st>>>(x, ldx, delta, ldd, mask, Mm, Dd);
}
static inline void fillf(hipStream_t st, float* p, float v, int n) {
  fill_kernel<<<(n + 255) / 256, 256, 0, st>>>(p, v, n);
}

extern "C" void kernel_launch(void* const* d_in, const int* in_sizes, int n_in,
                              void* d_out, int out_size, void* d_ws, size_t ws_size,
                              hipStream_t stream) {
  (void)in_sizes; (void)n_in; (void)out_size; (void)ws_size;
  const int*   seq      = (const int*)  d_in[0];
  const float* atom14   = (const float*)d_in[1];
  const float* rot      = (const float*)d_in[2];
  const float* trans    = (const float*)d_in[3];
  const int*   res_m    = (const int*)  d_in[4];
  const int*   mgm_m    = (const int*)  d_in[5];
  const int*   noi_m    = (const int*)  d_in[6];
  const int*   a14ex    = (const int*)  d_in[7];
  const int*   ei       = (const int*)  d_in[8];
  const float* atom_tab = (const float*)d_in[9];
  const float* Wn1 = (const float*)d_in[10], *Wn2 = (const float*)d_in[11],
             *Wn3 = (const float*)d_in[12];
  const float* Wez1 = (const float*)d_in[13], *Wez2 = (const float*)d_in[14],
             *Wez3 = (const float*)d_in[15];
  const float* Wa1 = (const float*)d_in[16], *Wa2 = (const float*)d_in[17],
             *Wa3 = (const float*)d_in[18];
  const float* Wq = (const float*)d_in[19], *Wk = (const float*)d_in[20],
             *Wv = (const float*)d_in[21];
  const float* Wqp = (const float*)d_in[22], *Wkp = (const float*)d_in[23];
  const float* Wb = (const float*)d_in[24], *gamma = (const float*)d_in[25];
  const float* Wo = (const float*)d_in[26], *Wau = (const float*)d_in[27];
  const float* Wt1 = (const float*)d_in[28], *Wt2 = (const float*)d_in[29];
  const float* Wne = (const float*)d_in[30];
  const float* We1 = (const float*)d_in[31], *We2 = (const float*)d_in[32];
  const float* Wp1 = (const float*)d_in[33], *Wp2 = (const float*)d_in[34];
  const float* Wmu = (const float*)d_in[35], *Wlv = (const float*)d_in[36];
  float* out = (float*)d_out;

  // ---- workspace bump allocator (256B aligned) ----
  char* base = (char*)d_ws;
  size_t off = 0;
  auto allocb = [&](size_t nbytes) -> char* {
    char* p = base + off;
    off += (nbytes + 255) & ~(size_t)255;
    return p;
  };
  auto alloc = [&](size_t nf) -> float* { return (float*)allocb(nf * sizeof(float)); };

  float* bb4      = alloc((size_t)NRES * 12);
  float* bb_local = alloc((size_t)NRES * 12);
  float* res_f    = alloc(NRES);
  float* mgm_f    = alloc(NRES);
  float* noi_f    = alloc(NRES);
  float* em_f     = alloc(NEDGE);
  float* am_f     = alloc(NATOM);
  float* node_raw = alloc((size_t)NRP * 64);
  float* tA       = alloc((size_t)NRP * 512);
  float* tB       = alloc((size_t)NRP * 512);
  float* s        = alloc((size_t)NRP * CS);
  float* z        = alloc((size_t)NEDGE * CZ);
  float* ebufA    = alloc((size_t)NEDGE * 384);   // edge_raw(352) / ecat(384) / h2(256)
  float* ebufB    = alloc((size_t)NEDGE * 256);
  float* af       = alloc((size_t)NATOM * 32);
  float* acat     = alloc((size_t)NATOM * 288);   // also hosts atom_raw (stride 32)
  float* atmp     = alloc((size_t)NATOM * 32);
  float* atmp2    = alloc((size_t)NATOM * 32);
  float* qb       = alloc((size_t)NRP * CS);
  float* kb       = alloc((size_t)NRP * CS);
  float* vb       = alloc((size_t)NRP * CS);
  float* qpb      = alloc((size_t)NRP * 96);
  float* kpb      = alloc((size_t)NRP * 96);
  float* qg       = alloc((size_t)NRES * 96);
  float* kg       = alloc((size_t)NRES * 96);
  float* logits   = alloc((size_t)NEDGE * NH);
  float* mb       = alloc((size_t)NRES * NH);
  float* db       = alloc((size_t)NRES * NH);
  float* og       = alloc((size_t)NRES * 96);
  float* nodecat  = alloc((size_t)NRP * 384);

  // ---- pack all weights to bf16 WMMA fragments ----
  auto packw = [&](const float* W, int Ktrue, int Nw) -> __bf16* {
    int KB = (Ktrue + 31) / 32;
    size_t nel = (size_t)KB * Nw * 32;
    __bf16* p = (__bf16*)allocb(nel * sizeof(__bf16));
    int tot = KB * Nw * 2;
    pack_w_kernel<<<(tot + 255) / 256, 256, 0, stream>>>(W, p, Ktrue, Nw, KB);
    return p;
  };
  __bf16* pWn1 = packw(Wn1, 40, 512);
  __bf16* pWn2 = packw(Wn2, 512, 512);
  __bf16* pWn3 = packw(Wn3, 512, 256);
  __bf16* pWez1 = packw(Wez1, 340, 256);
  __bf16* pWez2 = packw(Wez2, 256, 256);
  __bf16* pWez3 = packw(Wez3, 256, 128);
  __bf16* pWa1 = packw(Wa1, 9, 32);
  __bf16* pWa2 = packw(Wa2, 32, 32);
  __bf16* pWa3 = packw(Wa3, 32, 16);
  __bf16* pWmu = packw(Wmu, 256, 256);
  __bf16* pWlv = packw(Wlv, 256, 256);
  __bf16 *pWq[NL], *pWk[NL], *pWv[NL], *pWqp[NL], *pWkp[NL], *pWo[NL], *pWau[NL],
         *pWt1[NL], *pWt2[NL], *pWne[NL], *pWe1[NL], *pWe2[NL], *pWp1[NL], *pWp2[NL];
  for (int l = 0; l < NL; ++l) {
    pWq[l]  = packw(Wq  + (size_t)l * 256 * 256, 256, 256);
    pWk[l]  = packw(Wk  + (size_t)l * 256 * 256, 256, 256);
    pWv[l]  = packw(Wv  + (size_t)l * 256 * 256, 256, 256);
    pWqp[l] = packw(Wqp + (size_t)l * 256 * 96, 256, 96);
    pWkp[l] = packw(Wkp + (size_t)l * 256 * 96, 256, 96);
    pWo[l]  = packw(Wo  + (size_t)l * 384 * 256, 384, 256);
    pWau[l] = packw(Wau + (size_t)l * 272 * 16, 272, 16);
    pWt1[l] = packw(Wt1 + (size_t)l * 256 * 256, 256, 256);
    pWt2[l] = packw(Wt2 + (size_t)l * 256 * 256, 256, 256);
    pWne[l] = packw(Wne + (size_t)l * 256 * 128, 256, 128);
    pWe1[l] = packw(We1 + (size_t)l * 384 * 256, 384, 256);
    pWe2[l] = packw(We2 + (size_t)l * 256 * 128, 256, 128);
    pWp1[l] = packw(Wp1 + (size_t)l * 16 * 16, 16, 16);
    pWp2[l] = packw(Wp2 + (size_t)l * 16 * 16, 16, 16);
  }

  const int T = 256;

  // ---------------- prep ----------------
  mask_kernel<<<(NRES + T - 1) / T, T, 0, stream>>>(res_m, mgm_m, noi_m,
                                                    res_f, mgm_f, noi_f, NRES);
  bb_kernel<<<(NRES + T - 1) / T, T, 0, stream>>>(atom14, rot, trans, bb4, bb_local, NRES);
  node_raw_kernel<<<(NRES + T - 1) / T, T, 0, stream>>>(atom14, rot, trans, seq,
                                                        res_f, mgm_f, node_raw, NRES);
  atom_prep_kernel<<<(NATOM + T - 1) / T, T, 0, stream>>>(seq, a14ex, mgm_f,
                                                          atom_tab, acat, am_f, NRES);
  edge_raw_kernel<<<(NEDGE + T - 1) / T, T, 0, stream>>>(ei, seq, res_m, noi_f, bb4,
                                                         bb_local, rot, trans, ebufA,
                                                         em_f, NEDGE);

  // node MLP: 40 -> 512 -> 512 -> 256, LN
  gemm(stream, node_raw, 64, pWn1, tA, NRES, 40, 512, 512, 0, 512, 1);
  gemm(stream, tA, 512, pWn2, tB, NRES, 512, 512, 512, 0, 512, 1);
  gemm(stream, tB, 512, pWn3, s, NRES, 512, 256, 256, 0, 256, 0);
  ln(stream, s, 256, nullptr, 0, nullptr, NRES, 256);

  // atom MLP: 9 -> 32 -> 32 -> 16, LN (atom_raw lives in acat, stride 32)
  gemm(stream, acat, 32, pWa1, atmp, NATOM, 9, 32, 32, 0, 32, 1);
  gemm(stream, atmp, 32, pWa2, atmp2, NATOM, 32, 32, 32, 0, 32, 1);
  gemm(stream, atmp2, 32, pWa3, af, NATOM, 32, 16, 32, 0, 32, 0);
  ln(stream, af, 32, nullptr, 0, nullptr, NATOM, 16);

  // edge MLP: 340 -> 256 -> 256 -> 128, LN
  gemm(stream, ebufA, 352, pWez1, ebufB, NEDGE, 340, 256, 256, 0, 256, 1);
  gemm(stream, ebufB, 256, pWez2, ebufA, NEDGE, 256, 256, 256, 0, 256, 1);
  gemm(stream, ebufA, 256, pWez3, z, NEDGE, 256, 128, 128, 0, 128, 0);
  ln(stream, z, 128, nullptr, 0, nullptr, NEDGE, 128);

  // ---------------- layers ----------------
  for (int l = 0; l < NL; ++l) {
    const float* Wb_l = Wb + (size_t)l * 128 * 4;
    const float* gm_l = gamma + (size_t)l * 4;

    gemm(stream, s, 256, pWq[l], qb, NRES, 256, 256, 256, 0, 256, 0);
    gemm(stream, s, 256, pWk[l], kb, NRES, 256, 256, 256, 0, 256, 0);
    gemm(stream, s, 256, pWv[l], vb, NRES, 256, 256, 256, 0, 256, 0);
    gemm(stream, s, 256, pWqp[l], qpb, NRES, 256, 96, 96, 0, 96, 0);
    gemm(stream, s, 256, pWkp[l], kpb, NRES, 256, 96, 96, 0, 96, 0);
    frame_points_kernel<<<(NRES*32 + T - 1)/T, T, 0, stream>>>(qpb, rot, trans, qg, NRES);
    frame_points_kernel<<<(NRES*32 + T - 1)/T, T, 0, stream>>>(kpb, rot, trans, kg, NRES);

    fillf(stream, mb, -1e9f, NRES * NH);
    fillf(stream, db, 0.f, NRES * NH);
    fillf(stream, nodecat, 0.f, NRES * 384);
    fillf(stream, og, 0.f, NRES * 96);
    logits_kernel<<<(NEDGE*NH + T - 1)/T, T, 0, stream>>>(qb, kb, z, Wb_l, gm_l, qg,
                                                          kg, em_f, ei, logits, mb, NEDGE);
    expsum_kernel<<<(NEDGE*NH + T - 1)/T, T, 0, stream>>>(mb, em_f, ei, logits, db, NEDGE);
    attn_accum_kernel<<<(NEDGE*NH + T - 1)/T, T, 0, stream>>>(logits, db, vb, kg, ei,
                                                              nodecat, og, NEDGE);
    finish_o_kernel<<<(NRES*32 + T - 1)/T, T, 0, stream>>>(og, rot, trans, nodecat, NRES);

    // s = LN(s + nodecat @ Wo)
    gemm(stream, nodecat, 384, pWo[l], tA, NRES, 384, 256, 256, 0, 256, 0);
    ln(stream, s, 256, tA, 256, nullptr, NRES, 256);

    // af = LN(af + [af, s] @ Wau)
    acat_kernel<<<((NATOM*288) + T - 1)/T, T, 0, stream>>>(af, s, acat, NRES);
    gemm(stream, acat, 288, pWau[l], atmp, NATOM, 272, 16, 32, 0, 32, 0);
    ln(stream, af, 32, atmp, 32, nullptr, NATOM, 16);

    // s = LN(s + relu(s@Wt1)@Wt2) * res_f
    gemm(stream, s, 256, pWt1[l], tA, NRES, 256, 256, 256, 0, 256, 1);
    gemm(stream, tA, 256, pWt2[l], tB, NRES, 256, 256, 256, 0, 256, 0);
    ln(stream, s, 256, tB, 256, res_f, NRES, 256);

    // edge update: z = LN(relu([z,h_dst,h_src]@We1)@We2) * em_f
    gemm(stream, s, 256, pWne[l], tA, NRES, 256, 128, 128, 0, 128, 0);
    ecat_kernel<<<((NEDGE*384) + T - 1)/T, T, 0, stream>>>(z, tA, ei, ebufA, NEDGE);
    gemm(stream, ebufA, 384, pWe1[l], ebufB, NEDGE, 384, 256, 256, 0, 256, 1);
    gemm(stream, ebufB, 256, pWe2[l], z, NEDGE, 256, 128, 128, 0, 128, 0);
    ln(stream, z, 128, nullptr, 0, em_f, NEDGE, 128);

    // af = LN(af + relu(af@Wp1)@Wp2) * am_f
    gemm(stream, af, 32, pWp1[l], atmp, NATOM, 16, 16, 32, 0, 32, 1);
    gemm(stream, atmp, 32, pWp2[l], atmp2, NATOM, 16, 16, 32, 0, 32, 0);
    ln(stream, af, 32, atmp2, 32, am_f, NATOM, 16);
  }

  // output: [s@Wmu, s@Wlv]  (N x 512)
  gemm(stream, s, 256, pWmu, out, NRES, 256, 256, 512, 0, 256, 0);
  gemm(stream, s, 256, pWlv, out, NRES, 256, 256, 512, 256, 256, 0);
}
